// SplitCausalSelfAttention_22462678958187
// MI455X (gfx1250) — matrix operand order repaired and checked by reading, not compiled, not run on hardware
//
#include <hip/hip_runtime.h>
#include <hip/hip_bf16.h>

typedef __bf16 bf16_t;
typedef __attribute__((ext_vector_type(16))) __bf16 v16bf;
typedef __attribute__((ext_vector_type(8)))  float  v8f;

#define B_   4
#define T_   2048
#define C_   1024
#define H_   16
#define D_   64
#define NTOK (B_ * T_)          // 8192
#define INV_SCALE 0.125f        // 1/sqrt(D)

// ---------------------------------------------------------------------------
// WMMA helpers (CDNA5 16x16x32 bf16, fp32 accumulate)
// ---------------------------------------------------------------------------
__device__ inline v8f wmma_bf(v16bf a, v16bf b, v8f c) {
  return __builtin_amdgcn_wmma_f32_16x16x32_bf16(false, a, false, b,
                                                 (short)0, c, false, false);
}

// Async global->LDS 16B copy (gfx1250 GLOBAL_LOAD_ASYNC_TO_LDS_B128).
// ldsoff = LDS byte offset (low 32 bits of the generic pointer), gptr = global.
__device__ inline void async_copy_b128(unsigned ldsoff, const void* gptr) {
  asm volatile("global_load_async_to_lds_b128 %0, %1, off"
               :: "v"(ldsoff), "v"(gptr)
               : "memory");
}
__device__ inline void wait_async0() {
  asm volatile("s_wait_asynccnt 0" ::: "memory");
}
__device__ inline unsigned lds_off(const void* p) {
  return (unsigned)(unsigned long long)p;   // flat LDS addr keeps offset in [31:0]
}

// A fragment: 16x32 bf16 tile, row-major source with leading dim ld.
// Contiguous k order used here (same as frag_b/frag_bT): lanes 0-15 row m=lane, K = 0..15; lanes 16-31 row
// m=lane-16, K = 16..31; two K values packed per VGPR in order.
__device__ inline v16bf frag_a(const bf16_t* p0, int ld) {
  const int lane = threadIdx.x & 31;
  const bf16_t* p = p0 + (lane & 15) * ld + ((lane & 16) ? 16 : 0);
  v16bf a;
#pragma unroll
  for (int i = 0; i < 8; ++i) a[i] = p[i];
#pragma unroll
  for (int i = 0; i < 8; ++i) a[8 + i] = p[8 + i];
  return a;
}

__device__ inline v16bf frag_a_scaled(const bf16_t* p0, int ld, float s) {
  const int lane = threadIdx.x & 31;
  const bf16_t* p = p0 + (lane & 15) * ld + ((lane & 16) ? 16 : 0);
  v16bf a;
#pragma unroll
  for (int i = 0; i < 8; ++i) a[i] = (bf16_t)((float)p[i] * s);
#pragma unroll
  for (int i = 0; i < 8; ++i) a[8 + i] = (bf16_t)((float)p[8 + i] * s);
  return a;
}

// B fragment: 32x16 bf16 tile, source stored row-major K x N, leading dim ld.
// Contiguous k order used here (matches frag_a): lanes 0-15 col n=lane, K=0..15; lanes 16-31 col n=lane-16,
// K=16..31 (two K values per VGPR in order).
__device__ inline v16bf frag_b(const bf16_t* p0, int ld) {
  const int lane = threadIdx.x & 31;
  const bf16_t* p = p0 + ((lane & 16) ? 16 : 0) * ld + (lane & 15);
  v16bf b;
#pragma unroll
  for (int i = 0; i < 16; ++i) b[i] = p[i * ld];
  return b;
}

// B fragment where B = M^T and M is stored row-major N x K (B[k][n] = M[n][k]).
__device__ inline v16bf frag_bT(const bf16_t* p0, int ld) {
  const int lane = threadIdx.x & 31;
  const bf16_t* p = p0 + (lane & 15) * ld + ((lane & 16) ? 16 : 0);
  v16bf b;
#pragma unroll
  for (int i = 0; i < 16; ++i) b[i] = p[i];
  return b;
}

// ---------------------------------------------------------------------------
// fp32 -> bf16 conversion
// ---------------------------------------------------------------------------
__global__ __launch_bounds__(256)
void cvt_f32_to_bf16(const float* __restrict__ src, bf16_t* __restrict__ dst,
                     int n) {
  int i = blockIdx.x * 256 + threadIdx.x;
  int stride = gridDim.x * 256;
  for (; i < n; i += stride) dst[i] = (bf16_t)src[i];
}

// ---------------------------------------------------------------------------
// GEMM: A[NTOK x C] (bf16, row-major) @ W[C x C] (bf16, row-major)
// MODE 0: store bf16, head-split layout (B,H,T,D)
// MODE 1: store fp32 row-major + bias
// Block tile 64(M) x 128(N), K-step 32.  8 waves, each owns 32x32 (2x2 WMMA).
// Double-buffered async global->LDS staging.
// ---------------------------------------------------------------------------
template <int MODE>
__global__ __launch_bounds__(256)
void gemm_bf16(const bf16_t* __restrict__ A, const bf16_t* __restrict__ W,
               bf16_t* __restrict__ outBF, float* __restrict__ outF32,
               const float* __restrict__ bias) {
  __shared__ bf16_t sA[2][64 * 32];
  __shared__ bf16_t sB[2][32 * 128];

  const int tid  = threadIdx.x;
  const int lane = tid & 31;
  const int w    = tid >> 5;
  const int wm   = w >> 2;       // 0..1
  const int wn   = w & 3;        // 0..3
  const int m0   = blockIdx.y * 64;
  const int n0   = blockIdx.x * 128;

  // per-thread async-copy coordinates
  const int arow = tid >> 2, acol = (tid & 3) * 8;    // A: one 16B chunk
  const int brow = tid >> 3, bcol = (tid & 7) * 16;   // B: two 16B chunks

  auto issue = [&](int k0, int buf) {
    async_copy_b128(lds_off(&sA[buf][arow * 32 + acol]),
                    &A[(long)(m0 + arow) * C_ + k0 + acol]);
    async_copy_b128(lds_off(&sB[buf][brow * 128 + bcol]),
                    &W[(long)(k0 + brow) * C_ + n0 + bcol]);
    async_copy_b128(lds_off(&sB[buf][brow * 128 + bcol + 8]),
                    &W[(long)(k0 + brow) * C_ + n0 + bcol + 8]);
  };

  const v8f vzero = {0.f, 0.f, 0.f, 0.f, 0.f, 0.f, 0.f, 0.f};
  v8f acc[2][2];
#pragma unroll
  for (int i = 0; i < 2; ++i)
#pragma unroll
    for (int j = 0; j < 2; ++j) acc[i][j] = vzero;

  issue(0, 0);
  wait_async0();
  __syncthreads();

  for (int k0 = 0; k0 < C_; k0 += 32) {
    const int cur = (k0 >> 5) & 1;
    if (k0 + 32 < C_) issue(k0 + 32, cur ^ 1);

    v16bf af[2], bfr[2];
#pragma unroll
    for (int mt = 0; mt < 2; ++mt)
      af[mt] = frag_a(&sA[cur][(wm * 32 + mt * 16) * 32], 32);
#pragma unroll
    for (int nt = 0; nt < 2; ++nt)
      bfr[nt] = frag_b(&sB[cur][wn * 32 + nt * 16], 128);
#pragma unroll
    for (int mt = 0; mt < 2; ++mt)
#pragma unroll
      for (int nt = 0; nt < 2; ++nt)
        acc[mt][nt] = wmma_bf(af[mt], bfr[nt], acc[mt][nt]);

    wait_async0();     // own prefetch landed
    __syncthreads();   // everyone done reading cur + all prefetches visible
  }

  const int half = lane >> 4;
  const int cn   = lane & 15;
#pragma unroll
  for (int mt = 0; mt < 2; ++mt)
#pragma unroll
    for (int nt = 0; nt < 2; ++nt)
#pragma unroll
      for (int j = 0; j < 8; ++j) {
        int r = m0 + wm * 32 + mt * 16 + j + 8 * half;
        int c = n0 + wn * 32 + nt * 16 + cn;
        float v = acc[mt][nt][j];
        if (MODE == 0) {
          int bb = r >> 11, t = r & (T_ - 1);
          int h = c >> 6, d = c & 63;
          outBF[((long)(bb * H_ + h) * T_ + t) * D_ + d] = (bf16_t)v;
        } else {
          outF32[(long)r * C_ + c] = v + bias[c];
        }
      }
}

// ---------------------------------------------------------------------------
// Flash-attention: Q,K,V in (B,H,T,D) bf16; Y out (B,T,C) bf16.
// Block = 8 waves, each wave owns 32 query rows of one (b,h); K/V 32-row tiles
// double-buffered in LDS via async copies; online softmax in registers.
// ---------------------------------------------------------------------------
__global__ __launch_bounds__(256)
void attn_flash(const bf16_t* __restrict__ Q, const bf16_t* __restrict__ K,
                const bf16_t* __restrict__ V, bf16_t* __restrict__ Y) {
  __shared__ bf16_t sK[2][32 * 64];
  __shared__ bf16_t sV[2][32 * 64];
  __shared__ bf16_t sP[8][32 * 32];

  const int tid  = threadIdx.x;
  const int lane = tid & 31;
  const int w    = tid >> 5;
  const int bh   = blockIdx.x >> 3;   // 0..63
  const int qblk = blockIdx.x & 7;    // 0..7
  const int b    = bh >> 4;
  const int h    = bh & 15;
  const int q0   = qblk * 256 + w * 32;

  // per-thread async-copy coordinates: 32x64 tile, one 16B chunk per thread
  const int krow = tid >> 3, kcol = (tid & 7) * 8;

  auto issue = [&](int kt, int buf) {
    const long base = ((long)bh * T_ + kt * 32 + krow) * D_ + kcol;
    async_copy_b128(lds_off(&sK[buf][krow * 64 + kcol]), &K[base]);
    async_copy_b128(lds_off(&sV[buf][krow * 64 + kcol]), &V[base]);
  };

  // Q fragments, pre-scaled by 1/sqrt(D) (exact power of two in bf16)
  v16bf aq[2][2];
#pragma unroll
  for (int mt = 0; mt < 2; ++mt)
#pragma unroll
    for (int kc = 0; kc < 2; ++kc)
      aq[mt][kc] = frag_a_scaled(
          Q + ((long)bh * T_ + q0 + mt * 16) * D_ + kc * 32, D_, INV_SCALE);

  const v8f vzero = {0.f, 0.f, 0.f, 0.f, 0.f, 0.f, 0.f, 0.f};
  v8f o[2][4];
#pragma unroll
  for (int mt = 0; mt < 2; ++mt)
#pragma unroll
    for (int nt = 0; nt < 4; ++nt) o[mt][nt] = vzero;

  float mrow[16], lrow[16];
#pragma unroll
  for (int i = 0; i < 16; ++i) { mrow[i] = -3.0e38f; lrow[i] = 0.f; }

  const int half = lane >> 4;
  const int cn   = lane & 15;
  const int ntiles = (qblk + 1) * 8;  // keys 0 .. qblk*256+255

  issue(0, 0);
  wait_async0();
  __syncthreads();

  for (int kt = 0; kt < ntiles; ++kt) {
    const int k0  = kt * 32;
    const int cur = kt & 1;
    if (kt + 1 < ntiles) issue(kt + 1, cur ^ 1);

    if (k0 <= q0 + 31) {
      // S = Q * K^T   (32x32 tile, 8 wmma)
      v8f s[2][2];
#pragma unroll
      for (int mt = 0; mt < 2; ++mt)
#pragma unroll
        for (int nt = 0; nt < 2; ++nt) s[mt][nt] = vzero;
#pragma unroll
      for (int kc = 0; kc < 2; ++kc) {
        v16bf bk[2];
#pragma unroll
        for (int nt = 0; nt < 2; ++nt)
          bk[nt] = frag_bT(&sK[cur][nt * 16 * 64 + kc * 32], 64);
#pragma unroll
        for (int mt = 0; mt < 2; ++mt)
#pragma unroll
          for (int nt = 0; nt < 2; ++nt)
            s[mt][nt] = wmma_bf(aq[mt][kc], bk[nt], s[mt][nt]);
      }

      // causal mask (only on diagonal-crossing tiles)
      if (k0 + 31 > q0) {
#pragma unroll
        for (int mt = 0; mt < 2; ++mt)
#pragma unroll
          for (int nt = 0; nt < 2; ++nt)
#pragma unroll
            for (int j = 0; j < 8; ++j) {
              int qr = q0 + mt * 16 + j + 8 * half;
              int kc2 = k0 + nt * 16 + cn;
              if (kc2 > qr) s[mt][nt][j] = -3.0e38f;
            }
      }

      // online softmax, row-wise (rows striped across 16-lane halves)
#pragma unroll
      for (int mt = 0; mt < 2; ++mt)
#pragma unroll
        for (int j = 0; j < 8; ++j) {
          const int idx = mt * 8 + j;
          float v = fmaxf(s[mt][0][j], s[mt][1][j]);
#pragma unroll
          for (int off = 8; off >= 1; off >>= 1)
            v = fmaxf(v, __shfl_xor(v, off, 16));
          float newm  = fmaxf(mrow[idx], v);
          float alpha = __expf(mrow[idx] - newm);
          float p0 = __expf(s[mt][0][j] - newm);
          float p1 = __expf(s[mt][1][j] - newm);
          s[mt][0][j] = p0;
          s[mt][1][j] = p1;
          float rs = p0 + p1;
#pragma unroll
          for (int off = 8; off >= 1; off >>= 1)
            rs += __shfl_xor(rs, off, 16);
          lrow[idx] = lrow[idx] * alpha + rs;
          mrow[idx] = newm;
#pragma unroll
          for (int nt = 0; nt < 4; ++nt) o[mt][nt][j] *= alpha;
        }

      // P (bf16) -> wave-private LDS, then O += P * V  (8 wmma)
      bf16_t* sp = &sP[w][0];
#pragma unroll
      for (int mt = 0; mt < 2; ++mt)
#pragma unroll
        for (int nt = 0; nt < 2; ++nt)
#pragma unroll
          for (int j = 0; j < 8; ++j)
            sp[(mt * 16 + j + 8 * half) * 32 + nt * 16 + cn] =
                (bf16_t)s[mt][nt][j];

      v16bf bv[4];
#pragma unroll
      for (int nt = 0; nt < 4; ++nt)
        bv[nt] = frag_b(&sV[cur][nt * 16], 64);
#pragma unroll
      for (int mt = 0; mt < 2; ++mt) {
        v16bf ap = frag_a(&sp[mt * 16 * 32], 32);
#pragma unroll
        for (int nt = 0; nt < 4; ++nt)
          o[mt][nt] = wmma_bf(ap, bv[nt], o[mt][nt]);
      }
    }

    wait_async0();     // own prefetch landed
    __syncthreads();   // all waves done with cur, prefetched tile visible
  }

  // normalize and store Y (B,T,C) with c = h*64 + d
#pragma unroll
  for (int mt = 0; mt < 2; ++mt)
#pragma unroll
    for (int j = 0; j < 8; ++j) {
      const int idx = mt * 8 + j;
      const float inv = 1.0f / lrow[idx];
      const int qr = q0 + mt * 16 + j + 8 * half;
      const long base = ((long)(b * T_ + qr)) * C_ + h * D_;
#pragma unroll
      for (int nt = 0; nt < 4; ++nt)
        Y[base + nt * 16 + cn] = (bf16_t)(o[mt][nt][j] * inv);
    }
}

// ---------------------------------------------------------------------------
// Host launcher
// ---------------------------------------------------------------------------
extern "C" void kernel_launch(void* const* d_in, const int* in_sizes, int n_in,
                              void* d_out, int out_size, void* d_ws,
                              size_t ws_size, hipStream_t stream) {
  const float* X  = (const float*)d_in[0];
  const float* Wq = (const float*)d_in[1];
  const float* Wk = (const float*)d_in[2];
  const float* Wv = (const float*)d_in[3];
  const float* Wo = (const float*)d_in[4];
  const float* bo = (const float*)d_in[5];
  float* out = (float*)d_out;

  const long NX = (long)NTOK * C_;  // 8388608
  const long NW = (long)C_ * C_;    // 1048576

  bf16_t* ws  = (bf16_t*)d_ws;
  bf16_t* Xb  = ws;
  bf16_t* Wqb = Xb + NX;
  bf16_t* Wkb = Wqb + NW;
  bf16_t* Wvb = Wkb + NW;
  bf16_t* Wob = Wvb + NW;
  bf16_t* Qb  = Wob + NW;
  bf16_t* Kb  = Qb + NX;
  bf16_t* Vb  = Kb + NX;
  bf16_t* Yb  = Vb + NX;

  // fp32 -> bf16
  cvt_f32_to_bf16<<<(int)((NX + 255) / 256), 256, 0, stream>>>(X, Xb, (int)NX);
  cvt_f32_to_bf16<<<(int)((NW + 255) / 256), 256, 0, stream>>>(Wq, Wqb, (int)NW);
  cvt_f32_to_bf16<<<(int)((NW + 255) / 256), 256, 0, stream>>>(Wk, Wkb, (int)NW);
  cvt_f32_to_bf16<<<(int)((NW + 255) / 256), 256, 0, stream>>>(Wv, Wvb, (int)NW);
  cvt_f32_to_bf16<<<(int)((NW + 255) / 256), 256, 0, stream>>>(Wo, Wob, (int)NW);

  // QKV projections (head-split bf16 outputs)
  dim3 ggrid(C_ / 128, NTOK / 64);
  gemm_bf16<0><<<ggrid, 256, 0, stream>>>(Xb, Wqb, Qb, nullptr, nullptr);
  gemm_bf16<0><<<ggrid, 256, 0, stream>>>(Xb, Wkb, Kb, nullptr, nullptr);
  gemm_bf16<0><<<ggrid, 256, 0, stream>>>(Xb, Wvb, Vb, nullptr, nullptr);

  // attention: B*H * (T/256) blocks
  attn_flash<<<B_ * H_ * (T_ / 256), 256, 0, stream>>>(Qb, Kb, Vb, Yb);

  // output projection + bias (fp32 out)
  gemm_bf16<1><<<ggrid, 256, 0, stream>>>(Yb, Wob, nullptr, out, bo);
}
